// MPNNEncoder_56092272885986
// MI455X (gfx1250) — compile-verified, matrix operand order checked by reading, not run on hardware
//
#include <hip/hip_runtime.h>
#include <hip/hip_bf16.h>
#include <math.h>

typedef __attribute__((ext_vector_type(16))) _Float16 v16h;
typedef __attribute__((ext_vector_type(8)))  float    v8f;

#define SD 64
#define ZD 32
#define GD 16
#define INDIM 128
#define NROUNDS 6

#if defined(__HIP_DEVICE_COMPILE__) && __has_builtin(__builtin_amdgcn_global_load_async_to_lds_b128)
#define USE_ASYNC_LDS 1
typedef __attribute__((ext_vector_type(4))) int v4i_t;
typedef v4i_t __attribute__((address_space(1)))* gas1_v4i;
typedef v4i_t __attribute__((address_space(3)))* las3_v4i;
#endif

static __device__ __forceinline__ int laneid() { return (int)(threadIdx.x & 31); }

static __device__ __forceinline__ v8f wmma_step(v16h a, v16h b, v8f c) {
  return __builtin_amdgcn_wmma_f32_16x16x32_f16(false, a, false, b, (short)0, c, false, false);
}

static __device__ __forceinline__ float fast_sig(float x) {
  return __builtin_amdgcn_rcpf(1.f + __expf(-x));     // v_exp + v_rcp
}
static __device__ __forceinline__ float fast_tanh(float x) {
  return 2.f * __builtin_amdgcn_rcpf(1.f + __expf(-2.f * x)) - 1.f;
}

// Stage a contiguous block of nfloats f32 into LDS (raw f32 copy).
static __device__ __forceinline__ void stage32(const float* __restrict__ src, int nfloats,
                                               float* lds) {
  const int lane = laneid();
#ifdef USE_ASYNC_LDS
  for (int i = lane * 4; i < nfloats; i += 128)
    __builtin_amdgcn_global_load_async_to_lds_b128(
        (gas1_v4i)(src + i), (las3_v4i)(lds + i), 0, 0);
#else
  for (int i = lane * 4; i < nfloats; i += 128)
    *(float4*)(lds + i) = *(const float4*)(src + i);
#endif
}

static __device__ __forceinline__ void stage_wait() {
#ifdef USE_ASYNC_LDS
#if __has_builtin(__builtin_amdgcn_s_wait_asynccnt)
  __builtin_amdgcn_s_wait_asynccnt(0);
#else
  asm volatile("s_wait_asynccnt 0" ::: "memory");
#endif
#endif
}

// A-fragment (16x32) from f32 LDS tile [16][K]; cvt to f16 at fragment build.
static __device__ __forceinline__ v16h load_A32(const float* lds, int K, int kbase) {
  const int lane = laneid();
  const int m  = lane & 15;
  const int kh = (lane >> 4) << 3;   // 0 or 8
  const float* p = lds + m * K + kbase + kh;
  v16h a;
#pragma unroll
  for (int e = 0; e < 8; ++e) a[e] = (_Float16)p[e];          // 2x ds_load_b128
#pragma unroll
  for (int e = 0; e < 8; ++e) a[8 + e] = (_Float16)p[16 + e]; // 2x ds_load_b128
  return a;
}

// B-fragment from pre-packed layout: frag f occupies 512 halfs, lane-major.
static __device__ __forceinline__ v16h load_Bp(const _Float16* __restrict__ packed, int frag) {
  return ((const v16h*)packed)[frag * 32 + laneid()];   // 2 x global_load_b128
}

// ---------------- B-fragment pre-pack ----------------
// frag = (kbase/32)*(Ncols/16) + (nbase/16); within frag: lane*16 + e.
// Element (lane,e) holds B[k][n]: n = nbase+(lane&15),
// k = kbase + ((e>>3)<<4) + ((lane>>4)<<3) + (e&7).
__global__ void packB_kernel(const float* __restrict__ src, _Float16* __restrict__ dst,
                             int K, int Ncols, int transpose) {
  int i = blockIdx.x * blockDim.x + threadIdx.x;
  if (i >= K * Ncols) return;
  int e    = i & 15;
  int lane = (i >> 4) & 31;
  int frag = i >> 9;
  int nfn = Ncols >> 4;
  int kf = frag / nfn, nf = frag - kf * nfn;
  int n = nf * 16 + (lane & 15);
  int k = kf * 32 + ((e >> 3) << 4) + ((lane >> 4) << 3) + (e & 7);
  float v = transpose ? src[n * K + k] : src[k * Ncols + n];
  dst[i] = (_Float16)v;
}

__global__ void zero_kernel(float* __restrict__ p, int n) {
  int i = blockIdx.x * blockDim.x + threadIdx.x;
  if (i < n) p[i] = 0.f;
}

// ---------------- h = relu(x @ W_in + b_in) ----------------
__global__ void __launch_bounds__(128) input_kernel(const float* __restrict__ x,
    const _Float16* __restrict__ Wp, const float* __restrict__ b_in,
    float* __restrict__ h, int ntiles) {
  __shared__ float lds[4][16 * INDIM];
  const int wave = threadIdx.x >> 5;
  const int tile = blockIdx.x * 4 + wave;
  if (tile >= ntiles) return;
  float* lA = lds[wave];
  const int base = tile * 16;
  stage32(x + (size_t)base * INDIM, 16 * INDIM, lA);
  stage_wait();
  v16h a[4];
#pragma unroll
  for (int ks = 0; ks < 4; ++ks) a[ks] = load_A32(lA, INDIM, ks * 32);
  const int lane = laneid();
  const int n = lane & 15, mb = (lane >> 4) << 3;
#pragma unroll
  for (int nt = 0; nt < 4; ++nt) {
    v8f c = {};
#pragma unroll
    for (int ks = 0; ks < 4; ++ks)
      c = wmma_step(a[ks], load_Bp(Wp, ks * 4 + nt), c);
    float bv = b_in[nt * 16 + n];
#pragma unroll
    for (int j = 0; j < 8; ++j)
      h[(size_t)(base + mb + j) * SD + nt * 16 + n] = fmaxf(c[j] + bv, 0.f);
  }
}

// ---------------- m = relu(h @ Wm[r] + bm[r]) ----------------
__global__ void __launch_bounds__(128) msg_kernel(const float* __restrict__ h,
    const _Float16* __restrict__ Wp, const float* __restrict__ bm,
    float* __restrict__ m, int ntiles) {
  __shared__ float lds[4][16 * SD];
  const int wave = threadIdx.x >> 5;
  const int tile = blockIdx.x * 4 + wave;
  if (tile >= ntiles) return;
  float* lA = lds[wave];
  const int base = tile * 16;
  stage32(h + (size_t)base * SD, 16 * SD, lA);
  stage_wait();
  v16h a0 = load_A32(lA, SD, 0), a1 = load_A32(lA, SD, 32);
  const int lane = laneid();
  const int n = lane & 15, mb = (lane >> 4) << 3;
#pragma unroll
  for (int nt = 0; nt < 4; ++nt) {
    v8f c = {};
    c = wmma_step(a0, load_Bp(Wp, nt), c);
    c = wmma_step(a1, load_Bp(Wp, 4 + nt), c);
    float bv = bm[nt * 16 + n];
#pragma unroll
    for (int j = 0; j < 8; ++j)
      m[(size_t)(base + mb + j) * SD + nt * 16 + n] = fmaxf(c[j] + bv, 0.f);
  }
}

// ---------------- agg[dst] += m[src]  (edge scatter) ----------------
__global__ void scatter_kernel(const int* __restrict__ ei, const float* __restrict__ m,
                               float* __restrict__ agg, int E) {
  int gid = blockIdx.x * blockDim.x + threadIdx.x;
  if (gid >= E * 16) return;
  int e = gid >> 4;
  int q = (gid & 15) << 2;          // channel group of 4
  int s = ei[e];
  int d = ei[E + e];
  float4 v = *(const float4*)(m + (size_t)s * SD + q);
  float* ap = agg + (size_t)d * SD + q;
  unsafeAtomicAdd(ap + 0, v.x);
  unsafeAtomicAdd(ap + 1, v.y);
  unsafeAtomicAdd(ap + 2, v.z);
  unsafeAtomicAdd(ap + 3, v.w);
}

// ---------------- fused: msg = relu(agg@Wu+bu); h = GRU(msg, h) ----------------
__global__ void __launch_bounds__(128) update_kernel(
    const float* __restrict__ agg, float* __restrict__ h,
    const _Float16* __restrict__ Wup, const float* __restrict__ bu,
    const _Float16* __restrict__ Wihp, const float* __restrict__ b_ih,
    const _Float16* __restrict__ Whhp, const float* __restrict__ b_hh, int ntiles) {
  __shared__ float ldsA[4][16 * SD];
  __shared__ float ldsB[4][16 * SD];
  const int wave = threadIdx.x >> 5;
  const int tile = blockIdx.x * 4 + wave;
  if (tile >= ntiles) return;
  float* lA = ldsA[wave];
  float* lB = ldsB[wave];
  const int base = tile * 16;
  const int lane = laneid();
  const int n = lane & 15, mb = (lane >> 4) << 3;
  float* __restrict__ hT = h + (size_t)base * SD;

  // msg = relu(agg @ Wu + bu)
  stage32(agg + (size_t)base * SD, 16 * SD, lA);
  stage_wait();
  v16h a0 = load_A32(lA, SD, 0), a1 = load_A32(lA, SD, 32);
  float msg[4][8];
#pragma unroll
  for (int nt = 0; nt < 4; ++nt) {
    v8f c = {};
    c = wmma_step(a0, load_Bp(Wup, nt), c);
    c = wmma_step(a1, load_Bp(Wup, 4 + nt), c);
    float bv = bu[nt * 16 + n];
#pragma unroll
    for (int j = 0; j < 8; ++j) msg[nt][j] = fmaxf(c[j] + bv, 0.f);
  }
  // restage msg (C layout -> row-major f32 LDS), stage h tile
#pragma unroll
  for (int nt = 0; nt < 4; ++nt)
#pragma unroll
    for (int j = 0; j < 8; ++j)
      lA[(mb + j) * SD + nt * 16 + n] = msg[nt][j];
  stage32(hT, 16 * SD, lB);
  stage_wait();

  v16h am0 = load_A32(lA, SD, 0), am1 = load_A32(lA, SD, 32);
  v16h ah0 = load_A32(lB, SD, 0), ah1 = load_A32(lB, SD, 32);

  float hC[4][8];
#pragma unroll
  for (int nt = 0; nt < 4; ++nt)
#pragma unroll
    for (int j = 0; j < 8; ++j)
      hC[nt][j] = hT[(mb + j) * SD + nt * 16 + n];

  // W_ihT packed: 2 kf x 12 nf fragments; gate g cols live at nf = g*4 + nt.
#pragma unroll
  for (int nt = 0; nt < 4; ++nt) {
    v8f gxr = {}, gxz = {}, gxn = {}, ghr = {}, ghz = {}, ghn = {};
    gxr = wmma_step(am0, load_Bp(Wihp,      nt), gxr);
    gxr = wmma_step(am1, load_Bp(Wihp, 12 + nt), gxr);
    gxz = wmma_step(am0, load_Bp(Wihp,  4 + nt), gxz);
    gxz = wmma_step(am1, load_Bp(Wihp, 16 + nt), gxz);
    gxn = wmma_step(am0, load_Bp(Wihp,  8 + nt), gxn);
    gxn = wmma_step(am1, load_Bp(Wihp, 20 + nt), gxn);
    ghr = wmma_step(ah0, load_Bp(Whhp,      nt), ghr);
    ghr = wmma_step(ah1, load_Bp(Whhp, 12 + nt), ghr);
    ghz = wmma_step(ah0, load_Bp(Whhp,  4 + nt), ghz);
    ghz = wmma_step(ah1, load_Bp(Whhp, 16 + nt), ghz);
    ghn = wmma_step(ah0, load_Bp(Whhp,  8 + nt), ghn);
    ghn = wmma_step(ah1, load_Bp(Whhp, 20 + nt), ghn);
    const int col = nt * 16 + n;
    float bxr = b_ih[col], bxz = b_ih[SD + col], bxn = b_ih[2 * SD + col];
    float bhr = b_hh[col], bhz = b_hh[SD + col], bhn = b_hh[2 * SD + col];
#pragma unroll
    for (int j = 0; j < 8; ++j) {
      float r  = fast_sig(gxr[j] + bxr + ghr[j] + bhr);
      float z  = fast_sig(gxz[j] + bxz + ghz[j] + bhz);
      float nn = fast_tanh(gxn[j] + bxn + r * (ghn[j] + bhn));
      hT[(mb + j) * SD + col] = (1.f - z) * nn + z * hC[nt][j];
    }
  }
}

// ---------------- node heads: mu = h@W_mu+b_mu ; lv = h@W_lv+b_lv ----------------
__global__ void __launch_bounds__(128) heads_kernel(const float* __restrict__ h,
    const _Float16* __restrict__ Wmup, const float* __restrict__ b_mu,
    const _Float16* __restrict__ Wlvp, const float* __restrict__ b_lv,
    float* __restrict__ outMu, float* __restrict__ outLv, int ntiles) {
  __shared__ float lds[4][16 * SD];
  const int wave = threadIdx.x >> 5;
  const int tile = blockIdx.x * 4 + wave;
  if (tile >= ntiles) return;
  float* lA = lds[wave];
  const int base = tile * 16;
  stage32(h + (size_t)base * SD, 16 * SD, lA);
  stage_wait();
  v16h a0 = load_A32(lA, SD, 0), a1 = load_A32(lA, SD, 32);
  const int lane = laneid();
  const int n = lane & 15, mb = (lane >> 4) << 3;
#pragma unroll
  for (int nt = 0; nt < 2; ++nt) {
    v8f c = {};
    c = wmma_step(a0, load_Bp(Wmup, nt), c);
    c = wmma_step(a1, load_Bp(Wmup, 2 + nt), c);
    float bv = b_mu[nt * 16 + n];
#pragma unroll
    for (int j = 0; j < 8; ++j)
      outMu[(size_t)(base + mb + j) * ZD + nt * 16 + n] = c[j] + bv;
    v8f c2 = {};
    c2 = wmma_step(a0, load_Bp(Wlvp, nt), c2);
    c2 = wmma_step(a1, load_Bp(Wlvp, 2 + nt), c2);
    float bv2 = b_lv[nt * 16 + n];
#pragma unroll
    for (int j = 0; j < 8; ++j)
      outLv[(size_t)(base + mb + j) * ZD + nt * 16 + n] = c2[j] + bv2;
  }
}

// ---------------- column-sum of h for graph mean ----------------
__global__ void gsum_kernel(const float* __restrict__ h, float* __restrict__ gsum, int N) {
  __shared__ float acc[SD];
  int t = threadIdx.x;
  if (t < SD) acc[t] = 0.f;
  __syncthreads();
  int c = t & 63;
  float p = 0.f;
  for (int row = blockIdx.x * 4 + (t >> 6); row < N; row += gridDim.x * 4)
    p += h[(size_t)row * SD + c];
  atomicAdd(&acc[c], p);
  __syncthreads();
  if (t < SD) unsafeAtomicAdd(&gsum[t], acc[t]);
}

// ---------------- graph heads (tiny) ----------------
__global__ void graph_head_kernel(const float* __restrict__ gsum,
    const float* __restrict__ Wg_mu1, const float* __restrict__ bg_mu1,
    const float* __restrict__ Wg_mu2, const float* __restrict__ bg_mu2,
    const float* __restrict__ Wg_lv1, const float* __restrict__ bg_lv1,
    const float* __restrict__ Wg_lv2, const float* __restrict__ bg_lv2,
    float* __restrict__ out, float invN) {
  __shared__ float g[SD], t1[SD], t2[SD];
  int t = threadIdx.x;           // 64 threads
  g[t] = gsum[t] * invN;
  __syncthreads();
  float s1 = bg_mu1[t], s2 = bg_lv1[t];
  for (int k = 0; k < SD; ++k) {
    s1 += g[k] * Wg_mu1[k * SD + t];
    s2 += g[k] * Wg_lv1[k * SD + t];
  }
  t1[t] = fmaxf(s1, 0.f);
  t2[t] = fmaxf(s2, 0.f);
  __syncthreads();
  if (t < GD) {
    float a = bg_mu2[t], b = bg_lv2[t];
    for (int c2 = 0; c2 < SD; ++c2) {
      a += t1[c2] * Wg_mu2[c2 * GD + t];
      b += t2[c2] * Wg_lv2[c2 * GD + t];
    }
    out[t] = a;
    out[GD + t] = b;
  }
}

extern "C" void kernel_launch(void* const* d_in, const int* in_sizes, int n_in,
                              void* d_out, int out_size, void* d_ws, size_t ws_size,
                              hipStream_t stream) {
  const float* x     = (const float*)d_in[0];
  const int*   ei    = (const int*)d_in[1];
  const float* W_in  = (const float*)d_in[2];
  const float* b_in  = (const float*)d_in[3];
  const float* Wm    = (const float*)d_in[4];
  const float* bm    = (const float*)d_in[5];
  const float* Wu    = (const float*)d_in[6];
  const float* bu    = (const float*)d_in[7];
  const float* W_ih  = (const float*)d_in[8];
  const float* b_ih  = (const float*)d_in[9];
  const float* W_hh  = (const float*)d_in[10];
  const float* b_hh  = (const float*)d_in[11];
  const float* W_mu  = (const float*)d_in[12];
  const float* b_mu  = (const float*)d_in[13];
  const float* W_lv  = (const float*)d_in[14];
  const float* b_lv  = (const float*)d_in[15];
  const float* Wg_mu1 = (const float*)d_in[16];
  const float* bg_mu1 = (const float*)d_in[17];
  const float* Wg_mu2 = (const float*)d_in[18];
  const float* bg_mu2 = (const float*)d_in[19];
  const float* Wg_lv1 = (const float*)d_in[20];
  const float* bg_lv1 = (const float*)d_in[21];
  const float* Wg_lv2 = (const float*)d_in[22];
  const float* bg_lv2 = (const float*)d_in[23];

  const int N = in_sizes[0] / INDIM;
  const int E = in_sizes[1] / 2;

  // workspace carve (256B aligned)
  char* ws = (char*)d_ws;
  size_t off = 0;
  auto carve = [&](size_t bytes) -> void* {
    void* p = ws + off;
    off += (bytes + 255) & ~(size_t)255;
    return p;
  };
  float*    h     = (float*)carve((size_t)N * SD * 4);
  float*    m     = (float*)carve((size_t)N * SD * 4);
  float*    agg   = (float*)carve((size_t)N * SD * 4);
  float*    gsum  = (float*)carve(256);
  _Float16* Winp  = (_Float16*)carve((size_t)INDIM * SD * 2);
  _Float16* Wmp   = (_Float16*)carve((size_t)NROUNDS * SD * SD * 2);
  _Float16* Wup   = (_Float16*)carve((size_t)NROUNDS * SD * SD * 2);
  _Float16* Wihp  = (_Float16*)carve((size_t)SD * 3 * SD * 2);
  _Float16* Whhp  = (_Float16*)carve((size_t)SD * 3 * SD * 2);
  _Float16* Wmup  = (_Float16*)carve((size_t)SD * ZD * 2);
  _Float16* Wlvp  = (_Float16*)carve((size_t)SD * ZD * 2);

  // pack weights into per-lane WMMA B-fragment layout (f16)
  packB_kernel<<<(INDIM * SD + 255) / 256, 256, 0, stream>>>(W_in, Winp, INDIM, SD, 0);
  for (int r = 0; r < NROUNDS; ++r) {
    packB_kernel<<<(SD * SD + 255) / 256, 256, 0, stream>>>(
        Wm + (size_t)r * SD * SD, Wmp + (size_t)r * SD * SD, SD, SD, 0);
    packB_kernel<<<(SD * SD + 255) / 256, 256, 0, stream>>>(
        Wu + (size_t)r * SD * SD, Wup + (size_t)r * SD * SD, SD, SD, 0);
  }
  packB_kernel<<<(SD * 3 * SD + 255) / 256, 256, 0, stream>>>(W_ih, Wihp, SD, 3 * SD, 1);
  packB_kernel<<<(SD * 3 * SD + 255) / 256, 256, 0, stream>>>(W_hh, Whhp, SD, 3 * SD, 1);
  packB_kernel<<<(SD * ZD + 255) / 256, 256, 0, stream>>>(W_mu, Wmup, SD, ZD, 0);
  packB_kernel<<<(SD * ZD + 255) / 256, 256, 0, stream>>>(W_lv, Wlvp, SD, ZD, 0);

  const int ntiles = N / 16;                 // N = 100000 -> 6250 full tiles
  const int mm_blocks = (ntiles + 3) / 4;

  input_kernel<<<mm_blocks, 128, 0, stream>>>(x, Winp, b_in, h, ntiles);

  const int scat_threads = E * 16;
  for (int r = 0; r < NROUNDS; ++r) {
    msg_kernel<<<mm_blocks, 128, 0, stream>>>(h, Wmp + (size_t)r * SD * SD,
                                              bm + (size_t)r * SD, m, ntiles);
    zero_kernel<<<(N * SD + 255) / 256, 256, 0, stream>>>(agg, N * SD);
    scatter_kernel<<<(scat_threads + 255) / 256, 256, 0, stream>>>(ei, m, agg, E);
    update_kernel<<<mm_blocks, 128, 0, stream>>>(agg, h, Wup + (size_t)r * SD * SD,
                                                 bu + (size_t)r * SD,
                                                 Wihp, b_ih, Whhp, b_hh, ntiles);
  }

  float* outMu = (float*)d_out;
  float* outLv = outMu + (size_t)N * ZD;
  float* outG  = outLv + (size_t)N * ZD;
  heads_kernel<<<mm_blocks, 128, 0, stream>>>(h, Wmup, b_mu, Wlvp, b_lv, outMu, outLv, ntiles);
  zero_kernel<<<1, 64, 0, stream>>>(gsum, SD);
  gsum_kernel<<<256, 256, 0, stream>>>(h, gsum, N);
  graph_head_kernel<<<1, 64, 0, stream>>>(gsum, Wg_mu1, bg_mu1, Wg_mu2, bg_mu2,
                                          Wg_lv1, bg_lv1, Wg_lv2, bg_lv2,
                                          outG, 1.f / (float)N);
}
